// CoeffHeadKAN_53068615910215
// MI455X (gfx1250) — compile-verified, matrix operand order checked by reading, not run on hardware
//
#include <hip/hip_runtime.h>
#include <math.h>

// ---------------------------------------------------------------------------
// CoeffHeadKAN fused kernel for gfx1250 (MI455X, wave32, WMMA)
//
// h   = silu(x) @ base_w^T + bases @ spline_w^T   == [f | bases] @ W^T
//   M = 32768 (E), N = 512 (HID), K = 384 + 384*19 = 7680
// out = tanh(h) @ lin_w^T + lin_b                 (K = 512, N = 64)
//
// bf16 WMMA (v_wmma_f32_16x16x32_bf16), f32 accumulation. Features are
// generated on the fly per 32-row tile (never materialized in HBM).
// Double-buffered feature tiles -> 1 barrier per k-step; branchless spline.
// ---------------------------------------------------------------------------

#define E_ROWS   32768
#define IN_DIM   384
#define HID_DIM  512
#define NK_DIM   64
#define COEF     19
#define K1       (IN_DIM + IN_DIM * COEF)   // 7680
#define KT1      (K1 / 32)                  // 240 k-tiles
#define NT1      (HID_DIM / 16)             // 32 n-tiles
#define KT2      (HID_DIM / 32)             // 16
#define NT2      (NK_DIM / 16)              // 4
#define W_DWORDS ((long)KT1 * NT1 * 256)    // fragment dwords for big GEMM B
#define L_DWORDS ((long)KT2 * NT2 * 256)    // fragment dwords for head B

typedef __attribute__((ext_vector_type(16))) __bf16 v16bf;
typedef __attribute__((ext_vector_type(8)))  float  v8f;

union Frag {
    v16bf v;
    unsigned int u[8];
};

static __device__ __forceinline__ unsigned short f2bf(float f) {
    unsigned int u = __float_as_uint(f);
    u = u + 0x7FFFu + ((u >> 16) & 1u);      // round-to-nearest-even
    return (unsigned short)(u >> 16);
}
static __device__ __forceinline__ float bf2f(unsigned short s) {
    return __uint_as_float(((unsigned int)s) << 16);
}
static __device__ __forceinline__ v8f v8f_zero() {
    v8f x;
#pragma unroll
    for (int i = 0; i < 8; ++i) x[i] = 0.0f;
    return x;
}

// Branchless cardinal cubic B-spline on [0,4] (truncated-power basis).
// Equals Cox-de Boor N_{c,3} on the uniform grid with u = (x - grid[c]) / h.
// Clamping u to [0,4] yields exact 0 outside the support (B3(0)=B3(4)=0).
static __device__ __forceinline__ float bspline3(float u) {
    u = fminf(fmaxf(u, 0.0f), 4.0f);
    float r1 = fmaxf(u - 1.0f, 0.0f);
    float r2 = fmaxf(u - 2.0f, 0.0f);
    float r3 = fmaxf(u - 3.0f, 0.0f);
    float s = u * u * u;
    s = fmaf(r1 * r1, -4.0f * r1, s);
    s = fmaf(r2 * r2,  6.0f * r2, s);
    s = fmaf(r3 * r3, -4.0f * r3, s);
    return s * (1.0f / 6.0f);
}

// Cooperative 32x32 bf16 feature tile: rows wave + j*8, column = lane.
// k = kt*32 + lane; silu branch is wave-uniform since 384 % 32 == 0.
static __device__ __forceinline__ void compute_feat_tile(
    unsigned int (*at)[16], const unsigned short (*xs)[IN_DIM],
    int kt, int wave, int lane) {
    int k = kt * 32 + lane;
    if (k < IN_DIM) {
#pragma unroll
        for (int j = 0; j < 4; ++j) {
            int m = wave + j * 8;
            float xv = bf2f(xs[m][k]);
            float f = xv * (1.0f / (1.0f + expf(-xv)));   // silu
            ((unsigned short*)at)[m * 32 + lane] = f2bf(f);
        }
    } else {
        int jj = k - IN_DIM;
        int i  = jj / COEF;
        int c  = jj - i * COEF;
        float cf = (float)c;
#pragma unroll
        for (int j = 0; j < 4; ++j) {
            int m = wave + j * 8;
            float xv = bf2f(xs[m][i]);
            // u = (x - grid[c]) / h, grid[c] = (c-3)*h - 1, h = 1/8
            float u = (xv + 1.0f) * 8.0f - cf + 3.0f;
            ((unsigned short*)at)[m * 32 + lane] = f2bf(bspline3(u));
        }
    }
}

// ---------------------------------------------------------------------------
// Pack [f|bases] weight matrix W (logical K1 x HID, B-operand) into bf16
// WMMA fragments: dword index = ((kt*NT1 + nt)*32 + lane)*8 + v.
// 16-bit B 32x16 striping: n = lane%16, half = lane/16, kk = g*16+half*8+vv*2,
// dword = {bf16 B[k+1][n], bf16 B[k][n]}.
// ---------------------------------------------------------------------------
__global__ void pack_w_kernel(const float* __restrict__ base_w,     // [512,384]
                              const float* __restrict__ spline_w,   // [512,384*19]
                              unsigned int* __restrict__ Wp) {
    int idx = blockIdx.x * 256 + threadIdx.x;          // < KT1*NT1*256
    int v    = idx & 7;
    int lane = (idx >> 3) & 31;
    int tile = idx >> 8;                               // kt*NT1 + nt
    int nt   = tile & (NT1 - 1);
    int kt   = tile >> 5;
    int n    = nt * 16 + (lane & 15);
    int half = lane >> 4;
    int kk   = (v >> 2) * 16 + half * 8 + (v & 3) * 2;
    int k0   = kt * 32 + kk;

    float w0, w1;
    if (k0 < IN_DIM)          w0 = base_w[n * IN_DIM + k0];
    else                      w0 = spline_w[n * (IN_DIM * COEF) + (k0 - IN_DIM)];
    int k1 = k0 + 1;
    if (k1 < IN_DIM)          w1 = base_w[n * IN_DIM + k1];
    else                      w1 = spline_w[n * (IN_DIM * COEF) + (k1 - IN_DIM)];

    Wp[idx] = (unsigned int)f2bf(w0) | ((unsigned int)f2bf(w1) << 16);
}

// lin_w [64, 512] -> B operand (K=512, N=64), same fragment layout.
__global__ void pack_lin_kernel(const float* __restrict__ lin_w,
                                unsigned int* __restrict__ Lp) {
    int idx = blockIdx.x * 256 + threadIdx.x;          // < KT2*NT2*256
    int v    = idx & 7;
    int lane = (idx >> 3) & 31;
    int tile = idx >> 8;                               // kt*NT2 + nt
    int nt   = tile & (NT2 - 1);
    int kt   = tile >> 2;
    int n    = nt * 16 + (lane & 15);
    int half = lane >> 4;
    int kk   = (v >> 2) * 16 + half * 8 + (v & 3) * 2;
    int k0   = kt * 32 + kk;

    float w0 = lin_w[n * HID_DIM + k0];
    float w1 = lin_w[n * HID_DIM + k0 + 1];
    Lp[idx] = (unsigned int)f2bf(w0) | ((unsigned int)f2bf(w1) << 16);
}

// ---------------------------------------------------------------------------
// Fused KAN kernel: 32 rows per block, 8 waves; wave w owns N columns
// [w*64, w*64+64) in GEMM1 and one 16x16 output tile in GEMM2.
// ---------------------------------------------------------------------------
__global__ __launch_bounds__(256) void kan_fused_kernel(
    const float* __restrict__ z,        // [E,128]
    const float* __restrict__ msrc,     // [E,128]
    const float* __restrict__ mdst,     // [E,128]
    const unsigned int* __restrict__ Wp,
    const unsigned int* __restrict__ Lp,
    const float* __restrict__ lin_b,    // [64]
    float* __restrict__ out) {          // [E,64]

    __shared__ unsigned short xs[32][IN_DIM];     // tanh(x) bf16, 24 KB
    __shared__ unsigned int   atile[2][32][16];   // double-buffered A tile, 4 KB
    __shared__ unsigned int   hs[32][256];        // 32x512 bf16 h, 32 KB

    const int tid  = threadIdx.x;
    const int lane = tid & 31;
    const int wave = tid >> 5;
    const int rowbase = blockIdx.x * 32;
    const int half = lane >> 4;
    const int mlo  = lane & 15;

    // ---- Phase 1: load + tanh-squash 32 rows of concat(z, msrc, mdst) ----
    // Wave owns 4 rows; lanes stride columns (coalesced, branch folds at t).
#pragma unroll
    for (int rr = 0; rr < 4; ++rr) {
        int m = wave * 4 + rr;
        int r = rowbase + m;
#pragma unroll
        for (int t = 0; t < 12; ++t) {
            int c = lane + t * 32;
            float raw = (c < 128) ? z[r * 128 + c]
                      : (c < 256) ? msrc[r * 128 + (c - 128)]
                                  : mdst[r * 128 + (c - 256)];
            xs[m][c] = f2bf(tanhf(raw));
        }
    }
    __syncthreads();

    v8f acc[2][4];
#pragma unroll
    for (int i = 0; i < 2; ++i)
#pragma unroll
        for (int j = 0; j < 4; ++j) acc[i][j] = v8f_zero();

    const int nbase_tile = wave * 4;              // first of 4 n-tiles owned

    // ---- Phase 2: GEMM1 over K = 7680, features generated on the fly ----
    compute_feat_tile(atile[0], xs, 0, wave, lane);
    __syncthreads();

    for (int kt = 0; kt < KT1; ++kt) {
        const int cur = kt & 1;

        // Issue all 4 B fragments (8x global_load_b128) before any wait.
        Frag b[4];
        const unsigned int* bp =
            Wp + ((kt * NT1 + nbase_tile) << 8) + lane * 8;
#pragma unroll
        for (int nt = 0; nt < 4; ++nt)
#pragma unroll
            for (int v = 0; v < 8; ++v) b[nt].u[v] = bp[(nt << 8) + v];

        // A fragments (16-bit 16x32 striping) for both 16-row m-tiles.
        Frag a0, a1;
#pragma unroll
        for (int v = 0; v < 8; ++v) {
            int d = (v >> 2) * 8 + half * 4 + (v & 3);
            a0.u[v] = atile[cur][mlo][d];
            a1.u[v] = atile[cur][16 + mlo][d];
        }

        // Overlap next feature tile with in-flight loads / WMMAs.
        if (kt + 1 < KT1)
            compute_feat_tile(atile[cur ^ 1], xs, kt + 1, wave, lane);

#pragma unroll
        for (int nt = 0; nt < 4; ++nt) {
            acc[0][nt] = __builtin_amdgcn_wmma_f32_16x16x32_bf16(
                false, a0.v, false, b[nt].v, (short)0, acc[0][nt], false, false);
            acc[1][nt] = __builtin_amdgcn_wmma_f32_16x16x32_bf16(
                false, a1.v, false, b[nt].v, (short)0, acc[1][nt], false, false);
        }
        __syncthreads();   // one barrier per k-step (double-buffered atile)
    }

    // ---- Phase 3: h = tanh(acc) -> LDS (bf16) ----
#pragma unroll
    for (int mt = 0; mt < 2; ++mt)
#pragma unroll
        for (int nt = 0; nt < 4; ++nt)
#pragma unroll
            for (int r = 0; r < 8; ++r) {
                int m = mt * 16 + half * 8 + r;           // C/D striping
                int n = wave * 64 + nt * 16 + mlo;
                ((unsigned short*)hs)[m * HID_DIM + n] = f2bf(tanhf(acc[mt][nt][r]));
            }
    __syncthreads();

    // ---- Phase 4: out = h @ lin_w^T + b ; one 16x16 tile per wave ----
    const int mt2 = wave >> 2;
    const int nt2 = wave & 3;
    v8f o = v8f_zero();
    for (int kt = 0; kt < KT2; ++kt) {
        Frag a, b;
#pragma unroll
        for (int v = 0; v < 8; ++v) {
            int d = kt * 16 + (v >> 2) * 8 + half * 4 + (v & 3);
            a.u[v] = hs[mt2 * 16 + mlo][d];
            b.u[v] = Lp[((kt * NT2 + nt2) << 8) + lane * 8 + v];
        }
        o = __builtin_amdgcn_wmma_f32_16x16x32_bf16(
            false, a.v, false, b.v, (short)0, o, false, false);
    }

    int col  = nt2 * 16 + mlo;
    float bv = lin_b[col];
#pragma unroll
    for (int r = 0; r < 8; ++r) {
        int row = rowbase + mt2 * 16 + half * 8 + r;
        out[row * NK_DIM + col] = o[r] + bv;
    }
}

// ---------------------------------------------------------------------------
extern "C" void kernel_launch(void* const* d_in, const int* in_sizes, int n_in,
                              void* d_out, int out_size, void* d_ws, size_t ws_size,
                              hipStream_t stream) {
    const float* z        = (const float*)d_in[0];
    const float* mem_src  = (const float*)d_in[1];
    const float* mem_dst  = (const float*)d_in[2];
    const float* base_w   = (const float*)d_in[3];
    const float* spline_w = (const float*)d_in[4];
    const float* lin_w    = (const float*)d_in[5];
    const float* lin_b    = (const float*)d_in[6];
    float* out = (float*)d_out;

    unsigned int* Wp = (unsigned int*)d_ws;           // 7.86 MB
    unsigned int* Lp = Wp + W_DWORDS;                 // +64 KB

    pack_w_kernel<<<(int)(W_DWORDS / 256), 256, 0, stream>>>(base_w, spline_w, Wp);
    pack_lin_kernel<<<(int)(L_DWORDS / 256), 256, 0, stream>>>(lin_w, Lp);
    kan_fused_kernel<<<E_ROWS / 32, 256, 0, stream>>>(z, mem_src, mem_dst,
                                                      Wp, Lp, lin_b, out);
    (void)in_sizes; (void)n_in; (void)out_size; (void)ws_size;
}
